// ColBERT_77850577207387
// MI455X (gfx1250) — matrix operand order repaired and checked by reading, not compiled, not run on hardware
//
#include <hip/hip_runtime.h>
#include <hip/hip_bf16.h>

// ---------------------------------------------------------------------------
// ColBERT MaxSim on CDNA5 (gfx1250), wave32 + v_wmma_f32_16x16x32_bf16
// B=64, SQ=128, SD=1024, H=768, D=128
// ---------------------------------------------------------------------------

#define B_   64
#define SQ_  128
#define SD_  1024
#define H_   768
#define D_   128

typedef __attribute__((ext_vector_type(16))) __bf16 v16bf;
typedef __attribute__((ext_vector_type(8)))  float  v8f;

union FragU {
    uint4           q[2];   // 32 bytes
    unsigned short  us[16];
    v16bf           v;
};

// f32 -> bf16 via native fptrunc (gfx1250 has a native bf16 datapath)
__device__ __forceinline__ unsigned short f2bf(float f) {
    union { __bf16 b; unsigned short u; } x;
    x.b = (__bf16)f;
    return x.u;
}

// monotone f32 -> u32 encoding (order-preserving for atomicMax)
__device__ __forceinline__ unsigned enc_f32(float f) {
    union { float f; unsigned u; } x; x.f = f;
    return (x.u & 0x80000000u) ? ~x.u : (x.u | 0x80000000u);
}
__device__ __forceinline__ float dec_f32(unsigned e) {
    unsigned bits = (e & 0x80000000u) ? (e & 0x7FFFFFFFu) : ~e;
    union { unsigned u; float f; } x; x.u = bits;
    return x.f;
}

// Cross-lane XOR exchange via ds_swizzle group-of-32 mode (immediate mask,
// no address setup VALU, no LDS memory touched).  offset = xor<<10 | and 0x1F.
template <int XM>
__device__ __forceinline__ float swz_xor(float v) {
    int i = __builtin_amdgcn_ds_swizzle(__float_as_int(v), (XM << 10) | 0x1F);
    return __int_as_float(i);
}

// ---------------------------------------------------------------------------
// Kernel 0: pre-swizzle W (f32 [128][768]) into bf16 B-fragment order:
// Wb[dTile(8)][ks(24)][lane(32)][16 bf16]  (192 KB, done ONCE, L2-resident)
// lane: n = lane&15 (output dim within tile), half = lane>>4 selects K +0/+16
// ---------------------------------------------------------------------------
__global__ __launch_bounds__(256)
void prep_w_kernel(const float* __restrict__ Wm,
                   unsigned short* __restrict__ Wb) {
    const int idx = blockIdx.x * 256 + threadIdx.x;     // 8*24*32 = 6144 frags
    if (idx >= 8 * 24 * 32) return;
    const int lane  = idx & 31;
    const int ks    = (idx >> 5) % 24;
    const int dTile = idx / (24 * 32);
    const int n = lane & 15, half = lane >> 4;
    const float* src = &Wm[(size_t)(dTile * 16 + n) * H_ + ks * 32 + half * 16];
    unsigned short* dst = &Wb[(size_t)idx * 16];
    #pragma unroll
    for (int i = 0; i < 16; ++i) dst[i] = f2bf(src[i]);
}

// ---------------------------------------------------------------------------
// Kernel 1: project + L2-normalize.  One block = 16 tokens, 8 waves.
// Wave w computes output dims [16w, 16w+16) for all 16 tokens.
// emb[t][d] = normalize( (hidden[t][:] * mask[t]) @ W[d][:] )  stored bf16.
// ---------------------------------------------------------------------------
__global__ __launch_bounds__(256)
void encode_kernel(const float* __restrict__ hidden,
                   const int*   __restrict__ mask,
                   const unsigned short* __restrict__ Wb,  // pre-swizzled bf16
                   unsigned short* __restrict__ out /* bf16 [tokens][128] */) {
    __shared__ __align__(16) unsigned short aTile[16 * H_];   // 24 KB bf16
    __shared__ float s_sumsq[16];

    const int tid = threadIdx.x;
    const int t0  = blockIdx.x * 16;

    if (tid < 16) s_sumsq[tid] = 0.0f;

    // Stage masked hidden tile as bf16 in LDS (float4-vectorized)
    {
        const int nvec = 16 * H_ / 4;                    // 3072 float4 groups
        for (int i = tid; i < nvec; i += 256) {
            const int row = i / (H_ / 4);
            const int c4  = i - row * (H_ / 4);
            const float m = (float)mask[t0 + row];
            const float4 h = ((const float4*)hidden)[(size_t)(t0 + row) * (H_ / 4) + c4];
            unsigned short b0 = f2bf(h.x * m), b1 = f2bf(h.y * m);
            unsigned short b2 = f2bf(h.z * m), b3 = f2bf(h.w * m);
            ((uint2*)aTile)[i] = make_uint2((unsigned)b0 | ((unsigned)b1 << 16),
                                            (unsigned)b2 | ((unsigned)b3 << 16));
        }
    }
    __syncthreads();

    const int wave = tid >> 5;
    const int lane = tid & 31;
    const int n    = lane & 15;      // A-row (token)  and  B-col (output dim)
    const int half = lane >> 4;
    const int d0   = wave * 16;

    // Per-lane base into pre-swizzled W for this wave's d-tile
    const uint4* wbase = (const uint4*)&Wb[(((size_t)wave * 24) * 32 + lane) * 16];

    v8f acc = {};
    #pragma unroll 4
    for (int ks = 0; ks < H_ / 32; ++ks) {
        const int k0 = ks * 32;
        // A fragment (16x32 bf16): lane half selects K offset 0/16; elems 0..15 contiguous
        FragU af;
        {
            const uint4* pa = (const uint4*)&aTile[n * H_ + k0 + half * 16];
            af.q[0] = pa[0];          // K .. K+7
            af.q[1] = pa[1];          // K+8 .. K+15
        }
        // B fragment: pre-swizzled, contiguous 32B per lane, coalesced per wave
        FragU bf;
        {
            const uint4* wb = wbase + (size_t)ks * 32 * 2;   // 32 lanes * 2 uint4
            bf.q[0] = wb[0];
            bf.q[1] = wb[1];
        }
        acc = __builtin_amdgcn_wmma_f32_16x16x32_bf16(
                false, af.v, false, bf.v, (short)0, acc, false, false);
    }

    // Per-row sum of squares: row = j + 8*half; sum across the 16 lanes of the
    // half (XOR masks 1,2,4,8 never cross the 16-lane boundary).
    float ss[8];
    #pragma unroll
    for (int j = 0; j < 8; ++j) {
        float s = acc[j] * acc[j];
        s += swz_xor<1>(s);
        s += swz_xor<2>(s);
        s += swz_xor<4>(s);
        s += swz_xor<8>(s);
        ss[j] = s;
    }
    if (n == 0) {
        #pragma unroll
        for (int j = 0; j < 8; ++j)
            atomicAdd(&s_sumsq[j + half * 8], ss[j]);   // ds_add_f32
    }
    __syncthreads();

    // Normalize (norm over full D spans all 8 waves) and store bf16
    #pragma unroll
    for (int j = 0; j < 8; ++j) {
        const int row = j + half * 8;
        const float sc = 1.0f / fmaxf(sqrtf(s_sumsq[row]), 1e-12f);
        out[(size_t)(t0 + row) * D_ + d0 + n] = f2bf(acc[j] * sc);
    }
}

// ---------------------------------------------------------------------------
// Kernel 2: init per-(b,q) running max (encoded -inf == 0)
// ---------------------------------------------------------------------------
__global__ void init_kernel(unsigned* __restrict__ maxbuf) {
    int i = blockIdx.x * blockDim.x + threadIdx.x;
    if (i < B_ * SQ_) maxbuf[i] = 0u;
}

// ---------------------------------------------------------------------------
// Kernel 3: sim = q @ docT, running max over doc tokens.
// Grid = B * (SD/256). Block: 64KB LDS doc tile, 8 waves x 16-query tiles.
// ---------------------------------------------------------------------------
__global__ __launch_bounds__(256)
void sim_kernel(const unsigned short* __restrict__ qEmb,
                const unsigned short* __restrict__ dEmb,
                unsigned* __restrict__ maxbuf) {
    __shared__ __align__(16) unsigned short dT[256 * D_];   // 64 KB bf16

    const int tid   = threadIdx.x;
    const int b     = blockIdx.x >> 2;
    const int chunk = blockIdx.x & 3;
    const int t0    = chunk * 256;

    // Stage 256 doc embeddings (row-major, k-contiguous) into LDS
    {
        const uint4* src = (const uint4*)&dEmb[((size_t)b * SD_ + t0) * D_];
        uint4*       dst = (uint4*)dT;
        #pragma unroll 4
        for (int i = tid; i < 256 * D_ * 2 / 16; i += 256) dst[i] = src[i];
    }
    __syncthreads();

    const int wave = tid >> 5;
    const int lane = tid & 31;
    const int n    = lane & 15;
    const int half = lane >> 4;
    const int q0   = wave * 16;

    // Preload the 4 A fragments (this wave's 16 query rows, K = 0..127)
    FragU a[4];
    {
        const unsigned short* qrow = &qEmb[((size_t)b * SQ_ + q0 + n) * D_];
        #pragma unroll
        for (int ks = 0; ks < 4; ++ks) {
            const uint4* p = (const uint4*)&qrow[ks * 32 + half * 16];
            a[ks].q[0] = p[0];
            a[ks].q[1] = p[1];
        }
    }

    float rmax[8];
    #pragma unroll
    for (int j = 0; j < 8; ++j) rmax[j] = -3.402823466e38f;

    for (int dt = 0; dt < 16; ++dt) {
        v8f acc = {};
        const int tcol = dt * 16 + n;
        #pragma unroll
        for (int ks = 0; ks < 4; ++ks) {
            FragU bf;
            const uint4* pb = (const uint4*)&dT[tcol * D_ + ks * 32 + half * 16];
            bf.q[0] = pb[0];
            bf.q[1] = pb[1];
            acc = __builtin_amdgcn_wmma_f32_16x16x32_bf16(
                    false, a[ks].v, false, bf.v, (short)0, acc, false, false);
        }
        #pragma unroll
        for (int j = 0; j < 8; ++j) rmax[j] = fmaxf(rmax[j], acc[j]);
    }

    // Row max across the 16 doc columns held by this lane half
    #pragma unroll
    for (int j = 0; j < 8; ++j) {
        float v = rmax[j];
        v = fmaxf(v, swz_xor<1>(v));
        v = fmaxf(v, swz_xor<2>(v));
        v = fmaxf(v, swz_xor<4>(v));
        v = fmaxf(v, swz_xor<8>(v));
        rmax[j] = v;
    }
    if (n == 0) {
        const int rowbase = b * SQ_ + q0 + half * 8;
        #pragma unroll
        for (int j = 0; j < 8; ++j)
            atomicMax(&maxbuf[rowbase + j], enc_f32(rmax[j]));
    }
}

// ---------------------------------------------------------------------------
// Kernel 4: out[b] = sum over queries of decoded maxes
// ---------------------------------------------------------------------------
__global__ __launch_bounds__(128)
void finalize_kernel(const unsigned* __restrict__ maxbuf,
                     float* __restrict__ out) {
    __shared__ float red[4];
    const int b = blockIdx.x;
    const int t = threadIdx.x;
    float v = dec_f32(maxbuf[b * SQ_ + t]);
    v += swz_xor<16>(v);
    v += swz_xor<8>(v);
    v += swz_xor<4>(v);
    v += swz_xor<2>(v);
    v += swz_xor<1>(v);
    if ((t & 31) == 0) red[t >> 5] = v;
    __syncthreads();
    if (t == 0) out[b] = red[0] + red[1] + red[2] + red[3];
}

// ---------------------------------------------------------------------------
extern "C" void kernel_launch(void* const* d_in, const int* in_sizes, int n_in,
                              void* d_out, int out_size, void* d_ws, size_t ws_size,
                              hipStream_t stream) {
    (void)in_sizes; (void)n_in; (void)out_size; (void)ws_size;

    const float* qh = (const float*)d_in[0];   // [64,128,768] f32
    const int*   qm = (const int*)  d_in[1];   // [64,128] i32
    const float* dh = (const float*)d_in[2];   // [64,1024,768] f32
    const int*   dm = (const int*)  d_in[3];   // [64,1024] i32
    const float* Wm = (const float*)d_in[4];   // [128,768] f32
    float* out = (float*)d_out;                // [64] f32

    char* ws = (char*)d_ws;
    unsigned short* qEmb = (unsigned short*)ws;                              // 2 MB bf16
    unsigned short* dEmb = (unsigned short*)(ws + (size_t)2  * 1024 * 1024); // 16 MB bf16
    unsigned*       mbuf = (unsigned*)      (ws + (size_t)18 * 1024 * 1024); // 32 KB
    unsigned short* Wb   = (unsigned short*)(ws + (size_t)18 * 1024 * 1024
                                                + 64 * 1024);                // 192 KB

    // One-time W pre-swizzle to bf16 fragment layout
    prep_w_kernel<<<(8 * 24 * 32 + 255) / 256, 256, 0, stream>>>(Wm, Wb);

    // Project + normalize: queries (512 token-tiles), docs (4096 token-tiles)
    encode_kernel<<<B_ * SQ_ / 16, 256, 0, stream>>>(qh, qm, Wb, qEmb);
    encode_kernel<<<B_ * SD_ / 16, 256, 0, stream>>>(dh, dm, Wb, dEmb);

    init_kernel<<<(B_ * SQ_ + 255) / 256, 256, 0, stream>>>(mbuf);

    // MaxSim: 64 batches x 4 doc chunks of 256 tokens
    sim_kernel<<<B_ * (SD_ / 256), 256, 0, stream>>>(qEmb, dEmb, mbuf);

    finalize_kernel<<<B_, 128, 0, stream>>>(mbuf, out);
}
